// UniDirectionalRNN_32779190403213
// MI455X (gfx1250) — matrix operand order repaired and checked by reading, not compiled, not run on hardware
//
#include <hip/hip_runtime.h>
#include <hip/hip_bf16.h>
#include <stdint.h>

// Problem constants (from reference): B=64, T=2048, H=256, L=2, OUT=256
#define BATCH 64
#define SEQ   2048
#define HID   256
#define KDIM  512            // fused K = [x | h] = 2*H
#define NWG   16             // each WG owns 16 hidden columns
#define NTHREADS 256         // 8 wave32 waves

typedef __attribute__((ext_vector_type(16))) __bf16        v16bf;
typedef __attribute__((ext_vector_type(8)))  float         v8f;
typedef __attribute__((ext_vector_type(4)))  unsigned int  v4u;
typedef __attribute__((ext_vector_type(8)))  unsigned int  v8u;
typedef __attribute__((ext_vector_type(8)))  int           v8i;
typedef __attribute__((ext_vector_type(4)))  int           v4i;

// ---- LDS layout (dynamic shared, byte offsets) ----
// Wl   : [2][64][512] bf16  (both layers' weight slice, rows = 4 gate blocks x 16 cols)
// Abuf : [64][512]  bf16    (fused A = [x_t | h_prev], row stride 512)
// gates: [64][64]   f32     (this WG's 4 gate blocks x 16 cols)
// cst  : [2][64][16] f32    (persistent cell state)
// bias : [2][64]    f32     (b_ih + b_hh, WG slice)
#define LDS_W      0
#define LDS_ABUF   (2*64*KDIM*2)                 // 131072
#define LDS_GATES  (LDS_ABUF + BATCH*KDIM*2)     // +65536
#define LDS_CST    (LDS_GATES + BATCH*64*4)      // +16384
#define LDS_BIAS   (LDS_CST + 2*BATCH*16*4)      // +8192
#define LDS_TOTAL  (LDS_BIAS + 2*64*4)           // +512  => 221696 B (< 320KB WGP LDS)

// ------------------------------------------------------------------
// Tile loaders matching CDNA5 16-bit WMMA VGPR layouts (ISA 7.12.2).
// A (16x32, bf16): lane 0-15 -> rows, K per lane = {c..c+7, c+16..c+23}, c=8*(lane>=16)
// B (32x16, bf16): lane -> column n, K per lane = {c..c+7, c+16..c+23}, c=8*(lane>=16)
// Both reduce to two ds_load_b128 per lane from row-major LDS.
// ------------------------------------------------------------------
__device__ inline v16bf ld_a_tile(const __bf16* base, int stride, int row0, int k0, int lane) {
  int r = row0 + (lane & 15);
  int c = (lane >> 4) * 8;
  const unsigned int* p = (const unsigned int*)(base + r * stride + k0 + c);
  v4u q0 = *(const v4u*)p;
  v4u q1 = *(const v4u*)(p + 8);   // +16 bf16 elements
  v8u w; w.lo = q0; w.hi = q1;
  return __builtin_bit_cast(v16bf, w);
}

__device__ inline v16bf ld_b_tile(const __bf16* base, int stride, int n0, int k0, int lane) {
  int n = n0 + (lane & 15);
  int c = (lane >> 4) * 8;
  const unsigned int* p = (const unsigned int*)(base + n * stride + k0 + c);
  v4u q0 = *(const v4u*)p;
  v4u q1 = *(const v4u*)(p + 8);
  v8u w; w.lo = q0; w.hi = q1;
  return __builtin_bit_cast(v16bf, w);
}

__device__ inline v8f wmma_bf16(v16bf a, v16bf b, v8f c) {
  return __builtin_amdgcn_wmma_f32_16x16x32_bf16(false, a, false, b, (short)0, c, false, false);
}

// C/D layout: VGPR v, lanes 0-15: M=v, N=lane; lanes 16-31: M=v+8, N=lane-16
__device__ inline void st_gates(float* gates, int row0, int col0, v8f acc, int lane) {
  int col = col0 + (lane & 15);
  int rb  = row0 + ((lane >> 4) << 3);
#pragma unroll
  for (int v = 0; v < 8; ++v) gates[(rb + v) * 64 + col] = acc[v];
}

// ------------------------------------------------------------------
// TDM: async 2D tensor load, 64 rows x 256 bf16, global row stride 256,
// into LDS with pad_enable: after every 128 DWORDs (512B = one 256-elem
// bf16 row) skip 128 DWORDs -> interleaves into the 512-wide A buffer.
// D# built per ISA 8.3/8.4 (group0: count=1, lds_addr, global_addr, type=2;
// group1: data_size=2B, pad_interval=6(128dw), pad_amount=127(128dw),
// tensor_dim0=256, tensor_dim1=64, tile 256x64, dim0_stride=256).
// This toolchain's builtin is the 6-arg form:
//   (uint32x4 g0, int32x8 g1, int32x4 g2, int32x4 g3, int32x8 g4, i32 cpol)
// ------------------------------------------------------------------
__device__ inline void tdm_load_64x256_bf16(const void* gsrc, unsigned lds_off) {
  uint64_t ga = (uint64_t)(uintptr_t)gsrc;
  v4u g0 = { 1u,                                  // count=1, user descriptor
             lds_off,                             // lds_addr (bytes)
             (unsigned)ga,                        // global_addr[31:0]
             (unsigned)((ga >> 32) & 0x01FFFFFFu) | 0x80000000u }; // addr[56:32] | type=2
  v8i g1;
  g1[0] = (int)((1u << 16) | (1u << 20) | (6u << 22) | (127u << 25)); // data_size=2B, pad_en, interval, amount
  g1[1] = (int)((unsigned)HID << 16);    // tensor_dim0 = 256 (bits 63:48)
  g1[2] = (int)((unsigned)BATCH << 16);  // tensor_dim1 = 64  (bits 95:80)
  g1[3] = (int)((unsigned)HID << 16);    // tile_dim0   = 256 (bits 127:112)
  g1[4] = (int)BATCH;                    // tile_dim1   = 64
  g1[5] = (int)HID;                      // tensor_dim0_stride = 256 elements
  g1[6] = 0;
  g1[7] = 0;
  v4i g2 = {0, 0, 0, 0};
  v4i g3 = {0, 0, 0, 0};
  v8i g4 = {0, 0, 0, 0, 0, 0, 0, 0};
  __builtin_amdgcn_tensor_load_to_lds(g0, g1, g2, g3, g4, 0);
}

// ------------------------------------------------------------------
// Grid-wide barrier: cumulative-ticket arrive + spin (16 resident WGs,
// one per WGP; state zeroed by init kernel every launch).
// ------------------------------------------------------------------
__device__ inline void grid_sync(unsigned* bar, int& round) {
  __threadfence();
  __syncthreads();
  ++round;
  if (threadIdx.x == 0) {
    __hip_atomic_fetch_add(bar, 1u, __ATOMIC_ACQ_REL, __HIP_MEMORY_SCOPE_AGENT);
    unsigned target = (unsigned)round * (unsigned)NWG;
    while (__hip_atomic_load(bar, __ATOMIC_ACQUIRE, __HIP_MEMORY_SCOPE_AGENT) < target) {
      __builtin_amdgcn_s_sleep(1);
    }
  }
  __syncthreads();
}

__device__ inline float sigf(float x) { return 1.0f / (1.0f + __expf(-x)); }
__device__ inline float tanh_fast(float x) {
  float e = __expf(2.0f * x);
  return (e - 1.0f) / (e + 1.0f);
}

// One LSTM layer step for this WG's 16-column slice: gate GEMM (WMMA) + cell update.
__device__ inline void lstm_layer(const __bf16* Wl_l, const __bf16* Abuf, float* gates,
                                  float* cst_l, const float* bias_l, __bf16* hdst,
                                  int j, int g, int mp, int lane, int tid)
{
  v8f acc0 = {0.f,0.f,0.f,0.f,0.f,0.f,0.f,0.f};
  v8f acc1 = {0.f,0.f,0.f,0.f,0.f,0.f,0.f,0.f};
#pragma unroll 4
  for (int kt = 0; kt < KDIM / 32; ++kt) {
    v16bf bt = ld_b_tile(Wl_l, KDIM, g * 16, kt * 32, lane);   // shared across both M tiles
    v16bf a0 = ld_a_tile(Abuf, KDIM, mp * 16,       kt * 32, lane);
    v16bf a1 = ld_a_tile(Abuf, KDIM, (mp + 1) * 16, kt * 32, lane);
    acc0 = wmma_bf16(a0, bt, acc0);
    acc1 = wmma_bf16(a1, bt, acc1);
  }
  st_gates(gates, mp * 16,       g * 16, acc0, lane);
  st_gates(gates, (mp + 1) * 16, g * 16, acc1, lane);
  __syncthreads();
  // fused elementwise cell update for the WG's [64 x 16] slice
  for (int e = tid; e < BATCH * 16; e += NTHREADS) {
    int row = e >> 4, col = e & 15;
    float ig = gates[row * 64 +      col] + bias_l[     col];
    float fg = gates[row * 64 + 16 + col] + bias_l[16 + col];
    float gg = gates[row * 64 + 32 + col] + bias_l[32 + col];
    float og = gates[row * 64 + 48 + col] + bias_l[48 + col];
    float co = cst_l[row * 16 + col];
    float cn = sigf(fg) * co + sigf(ig) * tanh_fast(gg);
    float hn = sigf(og) * tanh_fast(cn);
    cst_l[row * 16 + col] = cn;
    hdst[row * HID + j * 16 + col] = (__bf16)hn;
  }
}

__global__ void __launch_bounds__(NTHREADS)
lstm_persistent(const float* __restrict__ x,
                const float* __restrict__ W_ih, const float* __restrict__ W_hh,
                const float* __restrict__ b_ih, const float* __restrict__ b_hh,
                const float* __restrict__ W_fc, const float* __restrict__ b_fc,
                __bf16* __restrict__ h0a, __bf16* __restrict__ h0b,
                __bf16* __restrict__ h1a, __bf16* __restrict__ h1b,
                unsigned* __restrict__ bar,
                float* __restrict__ out)
{
  extern __shared__ char smem[];
  __bf16* Wl    = (__bf16*)(smem + LDS_W);
  __bf16* Abuf  = (__bf16*)(smem + LDS_ABUF);
  float*  gates = (float*)(smem + LDS_GATES);
  float*  cst   = (float*)(smem + LDS_CST);
  float*  bias  = (float*)(smem + LDS_BIAS);
  __bf16* Wfc_s = (__bf16*)(smem + LDS_GATES);   // reused after the time loop

  const int tid  = threadIdx.x;
  const int lane = tid & 31;
  const int wave = tid >> 5;
  const int j    = blockIdx.x;                    // hidden-column slice [16j, 16j+16)
  const int g    = wave & 3;                      // gate block (i,f,g,o)
  const int mp   = (wave >> 2) * 2;               // M-tile pair base (0 or 2)

  // ---- one-time init: weight slice -> LDS bf16 (fused [W_ih | W_hh] along K) ----
  for (int idx = tid; idx < 2 * 64 * KDIM; idx += NTHREADS) {
    int l = idx >> 15, rem = idx & 32767;
    int n = rem >> 9, k = rem & 511;
    int gb = n >> 4, r = n & 15;
    int gr = gb * HID + j * 16 + r;               // global gate row
    float v = (k < HID) ? W_ih[(l * 4 * HID + gr) * HID + k]
                        : W_hh[(l * 4 * HID + gr) * HID + (k - HID)];
    Wl[(l * 64 + n) * KDIM + k] = (__bf16)v;
  }
  for (int idx = tid; idx < 2 * 64; idx += NTHREADS) {
    int l = idx >> 6, lc = idx & 63;
    int gb = lc >> 4, cc = lc & 15;
    int gr = gb * HID + j * 16 + cc;
    bias[idx] = b_ih[l * 4 * HID + gr] + b_hh[l * 4 * HID + gr];
  }
  for (int idx = tid; idx < 2 * BATCH * 16; idx += NTHREADS) cst[idx] = 0.f;
  __syncthreads();

  int round = 0;
  for (int t = 0; t < SEQ; ++t) {
    const int p = t & 1;
    __bf16* h0rd = p ? h0b : h0a;  __bf16* h0wr = p ? h0a : h0b;
    __bf16* h1rd = p ? h1b : h1a;  __bf16* h1wr = p ? h1a : h1b;

    // ======== layer 0: A = [x_t | h0_prev] ========
    if (wave == 0) tdm_load_64x256_bf16(h0rd, LDS_ABUF + 512);   // h half (async TDM)
    for (int i = tid; i < BATCH * HID; i += NTHREADS) {          // x half (cvt f32->bf16)
      int r = i >> 8, k = i & 255;
      Abuf[r * KDIM + k] = (__bf16)x[(r * SEQ + t) * HID + k];
    }
    if (wave == 0) __builtin_amdgcn_s_wait_tensorcnt(0);
    __syncthreads();
    lstm_layer(Wl, Abuf, gates, cst, bias, h0wr, j, g, mp, lane, tid);
    grid_sync(bar, round);

    // ======== layer 1: A = [h0_new | h1_prev] ========
    if (wave == 0) {
      tdm_load_64x256_bf16(h0wr, LDS_ABUF);
      tdm_load_64x256_bf16(h1rd, LDS_ABUF + 512);
      __builtin_amdgcn_s_wait_tensorcnt(0);
    }
    __syncthreads();
    lstm_layer(Wl + 64 * KDIM, Abuf, gates, cst + BATCH * 16, bias + 64, h1wr,
               j, g, mp, lane, tid);
    grid_sync(bar, round);
  }

  // ======== final FC on last h1: out[:, 16j..16j+16) = h1 @ W_fc^T + b_fc ========
  const int fin = ((SEQ - 1) & 1) ^ 1;              // index of last-written h1 buffer
  __bf16* h1fin = fin ? h1b : h1a;
  if (wave == 0) tdm_load_64x256_bf16(h1fin, LDS_ABUF);
  for (int idx = tid; idx < 16 * HID; idx += NTHREADS) {
    int r = idx >> 8, k = idx & 255;
    Wfc_s[r * HID + k] = (__bf16)W_fc[(j * 16 + r) * HID + k];
  }
  if (wave == 0) __builtin_amdgcn_s_wait_tensorcnt(0);
  __syncthreads();
  if (wave < 4) {
    v8f acc = {0.f,0.f,0.f,0.f,0.f,0.f,0.f,0.f};
#pragma unroll
    for (int kt = 0; kt < HID / 32; ++kt) {
      v16bf bt = ld_b_tile(Wfc_s, HID, 0, kt * 32, lane);
      v16bf a  = ld_a_tile(Abuf, KDIM, wave * 16, kt * 32, lane);
      acc = wmma_bf16(a, bt, acc);
    }
    int col = j * 16 + (lane & 15);
    float bb = b_fc[col];
    int rb = wave * 16 + ((lane >> 4) << 3);
#pragma unroll
    for (int v = 0; v < 8; ++v) out[(rb + v) * HID + col] = acc[v] + bb;
  }
}

__global__ void lstm_zero(unsigned* p, int n) {
  int i = blockIdx.x * blockDim.x + threadIdx.x;
  if (i < n) p[i] = 0u;
}

extern "C" void kernel_launch(void* const* d_in, const int* in_sizes, int n_in,
                              void* d_out, int out_size, void* d_ws, size_t ws_size,
                              hipStream_t stream) {
  (void)in_sizes; (void)n_in; (void)out_size; (void)ws_size;
  const float* x    = (const float*)d_in[0];
  const float* W_ih = (const float*)d_in[1];
  const float* W_hh = (const float*)d_in[2];
  const float* b_ih = (const float*)d_in[3];
  const float* b_hh = (const float*)d_in[4];
  const float* W_fc = (const float*)d_in[5];
  const float* b_fc = (const float*)d_in[6];

  // Workspace: [0,128) barrier; then 4 ping-pong bf16 h buffers of B*H each.
  char* ws = (char*)d_ws;
  unsigned* bar = (unsigned*)ws;
  __bf16* h0a = (__bf16*)(ws + 128);
  __bf16* h0b = h0a + BATCH * HID;
  __bf16* h1a = h0b + BATCH * HID;
  __bf16* h1b = h1a + BATCH * HID;

  // Re-zero barrier + h state every call (graph-replay safe, deterministic).
  int zeroN = (128 + 4 * BATCH * HID * 2) / 4;
  lstm_zero<<<(zeroN + 255) / 256, 256, 0, stream>>>((unsigned*)d_ws, zeroN);

  lstm_persistent<<<NWG, NTHREADS, LDS_TOTAL, stream>>>(
      x, W_ih, W_hh, b_ih, b_hh, W_fc, b_fc,
      h0a, h0b, h1a, h1b, bar, (float*)d_out);
}
